// AttentionMap_42812234006994
// MI455X (gfx1250) — matrix operand order repaired and checked by reading, not compiled, not run on hardware
//
#include <hip/hip_runtime.h>

// ---------------------------------------------------------------------------
// DANet spatial attention, fused flash-style, bf16 WMMA / f32 accumulate.
// B=16, C=512, H=W=48 -> HW=2304, d=64.
// Pipeline:
//   k0: convert Wq||Wk||Wv -> bf16 [640][512] + fused bias[640]   (workspace)
//   k1: projections via WMMA -> q,k [b][n][64] bf16, v [b][c][2304] bf16
//   k2: softmax row stats (online max/sumexp via WMMA logits)
//   k3: recompute logits -> P chunk in LDS -> v·P^T via WMMA, + gamma*out + x
// Hot loops are software-pipelined: fragments for step k+1 are fetched before
// the WMMA of step k issues.
// ---------------------------------------------------------------------------

typedef __bf16 bf16_t;
typedef __attribute__((ext_vector_type(16))) __bf16 v16bf;
typedef __attribute__((ext_vector_type(8)))  __bf16 v8bf;
typedef __attribute__((ext_vector_type(8)))  float  v8f;

#define NB 16
#define NC 512
#define NHW 2304
#define ND 64
#define NO 640   // 64 q + 64 k + 512 v output channels

__device__ __forceinline__ v16bf cat16(v8bf lo, v8bf hi) {
  return __builtin_shufflevector(lo, hi, 0,1,2,3,4,5,6,7,8,9,10,11,12,13,14,15);
}
__device__ __forceinline__ v16bf load_frag(const bf16_t* p0, const bf16_t* p1) {
  v8bf lo = *(const v8bf*)p0;
  v8bf hi = *(const v8bf*)p1;
  return cat16(lo, hi);
}
#define WMMA_BF16(A,B,C) \
  __builtin_amdgcn_wmma_f32_16x16x32_bf16(false,(A),false,(B),(short)0,(C),false,false)

// ---------------------------------------------------------------------------
// Kernel 0: pre-convert weights to bf16 [o][c] (o: 0-63 q, 64-127 k, 128-639 v)
// and gather biases into one array. Grid: 160 x 256 threads, 8 elems each.
// ---------------------------------------------------------------------------
__global__ __launch_bounds__(256)
void wcvt_kernel(const float* __restrict__ Wq, const float* __restrict__ bq,
                 const float* __restrict__ Wk, const float* __restrict__ bk,
                 const float* __restrict__ Wv, const float* __restrict__ bv,
                 bf16_t* __restrict__ Wb, float* __restrict__ bb)
{
  int g = blockIdx.x * 256 + threadIdx.x;
  int i = g * 8;                               // flat element index, 8 per thread
  int o = i >> 9;                              // /512
  int c = i & (NC - 1);
  const float* src;
  if (o < 64)        src = Wq + (size_t)o * NC + c;
  else if (o < 128)  src = Wk + (size_t)(o - 64) * NC + c;
  else               src = Wv + (size_t)(o - 128) * NC + c;
  float4 f0 = *(const float4*)(src);
  float4 f1 = *(const float4*)(src + 4);
  v8bf w;
  w[0]=(bf16_t)f0.x; w[1]=(bf16_t)f0.y; w[2]=(bf16_t)f0.z; w[3]=(bf16_t)f0.w;
  w[4]=(bf16_t)f1.x; w[5]=(bf16_t)f1.y; w[6]=(bf16_t)f1.z; w[7]=(bf16_t)f1.w;
  *(v8bf*)(Wb + (size_t)i) = w;
  if (g < NO)
    bb[g] = (g < 64) ? bq[g] : (g < 128) ? bk[g - 64] : bv[g - 128];
}

// ---------------------------------------------------------------------------
// Kernel 1: projections. Grid (144, 16): one 16-spatial tile per workgroup,
// all 640 output channels (8 waves x 5 out-tiles each). K-loop pipelined.
// ---------------------------------------------------------------------------
__global__ __launch_bounds__(256)
void proj_kernel(const float* __restrict__ x,
                 const bf16_t* __restrict__ Wb, const float* __restrict__ bb,
                 bf16_t* __restrict__ qws, bf16_t* __restrict__ kws,
                 bf16_t* __restrict__ vws)
{
  __shared__ bf16_t xs[16 * NC];            // x^T tile: [n_local][c], c contiguous
  const int b   = blockIdx.y;
  const int n0  = blockIdx.x * 16;
  const int tid = threadIdx.x;
  const float* xb = x + (size_t)b * NC * NHW;

  // stage + transpose + convert x tile: 2048 float4 loads across 256 threads
  #pragma unroll
  for (int kiter = 0; kiter < 8; ++kiter) {
    int idx = tid + kiter * 256;            // float4 index
    int c   = idx >> 2;
    int jj  = (idx & 3) * 4;
    float4 f = *(const float4*)(xb + (size_t)c * NHW + n0 + jj);
    xs[(jj + 0) * NC + c] = (bf16_t)f.x;
    xs[(jj + 1) * NC + c] = (bf16_t)f.y;
    xs[(jj + 2) * NC + c] = (bf16_t)f.z;
    xs[(jj + 3) * NC + c] = (bf16_t)f.w;
  }
  __syncthreads();

  const int wave = tid >> 5;
  const int lane = tid & 31;
  const int lrow = lane & 15;
  const int hf   = lane >> 4;
  const int kbA  = hf * 8;                  // unused: A frag also uses kbB (k = 16*hf + s)
  const int kbB  = hf * 16;                 // B-frag K base

  const bf16_t* wbase = Wb + (size_t)(wave * 16 + lrow) * NC + kbB; // tile i: +i*128*NC

  v8f acc[5];
  #pragma unroll
  for (int i = 0; i < 5; ++i) acc[i] = v8f{};

  // prologue: fragments for ct = 0
  v16bf bfrag[5];
  #pragma unroll
  for (int i = 0; i < 5; ++i) {
    const bf16_t* wp = wbase + (size_t)i * 128 * NC;
    bfrag[i] = load_frag(wp, wp + 8);
  }

  for (int ct = 0; ct < NC; ct += 32) {
    v16bf a = load_frag(&xs[lrow * NC + ct + kbB], &xs[lrow * NC + ct + kbB + 8]);
    // prefetch next K-step's B fragments before issuing WMMAs
    v16bf nb[5];
    if (ct + 32 < NC) {
      #pragma unroll
      for (int i = 0; i < 5; ++i) {
        const bf16_t* wp = wbase + (size_t)i * 128 * NC + ct + 32;
        nb[i] = load_frag(wp, wp + 8);
      }
    }
    #pragma unroll
    for (int i = 0; i < 5; ++i) acc[i] = WMMA_BF16(a, bfrag[i], acc[i]);
    if (ct + 32 < NC) {
      #pragma unroll
      for (int i = 0; i < 5; ++i) bfrag[i] = nb[i];
    }
  }

  #pragma unroll
  for (int i = 0; i < 5; ++i) {
    int o = (wave + 8 * i) * 16 + lrow;
    float bias = bb[o];
    #pragma unroll
    for (int r = 0; r < 8; ++r) {
      int n = n0 + r + 8 * hf;              // D row (f32 C/D layout)
      float val = acc[i][r] + bias;
      if (o < 64)
        qws[((size_t)b * NHW + n) * ND + o] = (bf16_t)val;
      else if (o < 128)
        kws[((size_t)b * NHW + n) * ND + (o - 64)] = (bf16_t)val;
      else
        vws[((size_t)b * NC + (o - 128)) * NHW + n] = (bf16_t)val;
    }
  }
}

// ---------------------------------------------------------------------------
// Kernel 2: per-row softmax stats (max, sumexp) over logits q[n,:]·k[m,:].
// Grid (144, 16): waves stride over 144 m-tiles with online (max,sum);
// k-fragment loads pipelined one tile ahead; then shuffle/LDS reduction.
// ---------------------------------------------------------------------------
__global__ __launch_bounds__(256)
void softmax_stats_kernel(const bf16_t* __restrict__ qws,
                          const bf16_t* __restrict__ kws,
                          float* __restrict__ Mg, float* __restrict__ Lg)
{
  __shared__ float redM[8][16];
  __shared__ float redS[8][16];
  const int b   = blockIdx.y;
  const int n0  = blockIdx.x * 16;
  const int tid = threadIdx.x;
  const int wave = tid >> 5, lane = tid & 31;
  const int lrow = lane & 15, hf = lane >> 4;
  const int kbA = hf * 8, kbB = hf * 16;

  const bf16_t* qr = qws + ((size_t)b * NHW + n0 + lrow) * ND;
  v16bf aq0 = load_frag(qr + kbB,      qr + kbB + 8);      // K = d 0..31
  v16bf aq1 = load_frag(qr + 32 + kbB, qr + 40 + kbB);     // K = d 32..63

  float lmax[8], lsum[8];
  #pragma unroll
  for (int r = 0; r < 8; ++r) { lmax[r] = -3.0e38f; lsum[r] = 0.f; }

  const bf16_t* kb0 = kws + (size_t)b * NHW * ND + kbB;
  // prologue for mt = wave
  const bf16_t* kr = kb0 + (size_t)(wave * 16 + lrow) * ND;
  v16bf b0 = load_frag(kr,      kr + 8);
  v16bf b1 = load_frag(kr + 32, kr + 40);

  for (int mt = wave; mt < NHW / 16; mt += 8) {
    v16bf nb0, nb1;
    if (mt + 8 < NHW / 16) {                 // prefetch next m-tile
      const bf16_t* kn = kb0 + (size_t)((mt + 8) * 16 + lrow) * ND;
      nb0 = load_frag(kn,      kn + 8);
      nb1 = load_frag(kn + 32, kn + 40);
    }
    v8f s = v8f{};
    s = WMMA_BF16(aq0, b0, s);
    s = WMMA_BF16(aq1, b1, s);
    #pragma unroll
    for (int r = 0; r < 8; ++r) {
      float v  = s[r];
      float nm = fmaxf(lmax[r], v);
      lsum[r]  = lsum[r] * __expf(lmax[r] - nm) + __expf(v - nm);
      lmax[r]  = nm;
    }
    if (mt + 8 < NHW / 16) { b0 = nb0; b1 = nb1; }
  }

  // reduce over the 16 lanes of each half (each half holds one row set)
  #pragma unroll
  for (int r = 0; r < 8; ++r) {
    #pragma unroll
    for (int off = 1; off < 16; off <<= 1) {
      float om = __shfl_xor(lmax[r], off, 32);
      float os = __shfl_xor(lsum[r], off, 32);
      float nm = fmaxf(lmax[r], om);
      lsum[r]  = lsum[r] * __expf(lmax[r] - nm) + os * __expf(om - nm);
      lmax[r]  = nm;
    }
  }
  if (lrow == 0) {                          // lanes 0 and 16 of each wave
    #pragma unroll
    for (int r = 0; r < 8; ++r) {
      redM[wave][r + 8 * hf] = lmax[r];
      redS[wave][r + 8 * hf] = lsum[r];
    }
  }
  __syncthreads();
  if (tid < 16) {
    float mx = -3.0e38f, sm = 0.f;
    #pragma unroll
    for (int w = 0; w < 8; ++w) {
      float om = redM[w][tid], os = redS[w][tid];
      float nm = fmaxf(mx, om);
      sm = sm * __expf(mx - nm) + os * __expf(om - nm);
      mx = nm;
    }
    Mg[(size_t)b * NHW + n0 + tid] = mx;
    Lg[(size_t)b * NHW + n0 + tid] = sm;
  }
}

// ---------------------------------------------------------------------------
// Kernel 3: out[:, n-tile] = v · P^T, fused with y = gamma*out + x.
// Grid (144, 16). Each iteration: 8 waves cooperatively produce a 16x256
// bf16 P chunk in LDS, then each wave runs the PV GEMM for its 64 channels
// (4 c-tiles x 8 K32 WMMAs, fragment loads pipelined one step ahead).
// ---------------------------------------------------------------------------
__global__ __launch_bounds__(256)
void attn_out_kernel(const float* __restrict__ x, const float* __restrict__ gamma_p,
                     const bf16_t* __restrict__ qws, const bf16_t* __restrict__ kws,
                     const bf16_t* __restrict__ vws,
                     const float* __restrict__ Mg, const float* __restrict__ Lg,
                     float* __restrict__ out)
{
  __shared__ bf16_t Pl[16 * 256];           // P chunk: [n_local][m_local]
  const int b   = blockIdx.y;
  const int n0  = blockIdx.x * 16;
  const int tid = threadIdx.x;
  const int wave = tid >> 5, lane = tid & 31;
  const int lrow = lane & 15, hf = lane >> 4;
  const int kbA = hf * 8, kbB = hf * 16;
  const float gamma = gamma_p[0];

  const bf16_t* qr = qws + ((size_t)b * NHW + n0 + lrow) * ND;
  v16bf aq0 = load_frag(qr + kbB,      qr + kbB + 8);
  v16bf aq1 = load_frag(qr + 32 + kbB, qr + 40 + kbB);

  float rowM[8], rowInvL[8];
  #pragma unroll
  for (int r = 0; r < 8; ++r) {
    int n = n0 + r + 8 * hf;
    rowM[r]    = Mg[(size_t)b * NHW + n];
    rowInvL[r] = 1.0f / Lg[(size_t)b * NHW + n];
  }

  v8f acc[4];
  #pragma unroll
  for (int i = 0; i < 4; ++i) acc[i] = v8f{};

  for (int it = 0; it < NHW / 256; ++it) {
    const int mbase = it * 256;
    // --- produce this wave's 16x32 slice of P ---
    #pragma unroll
    for (int sub = 0; sub < 2; ++sub) {
      int mt = mbase + wave * 32 + sub * 16;
      int m  = mt + lrow;
      const bf16_t* kr = kws + ((size_t)b * NHW + m) * ND;
      v16bf b0 = load_frag(kr + kbB,      kr + kbB + 8);
      v16bf b1 = load_frag(kr + 32 + kbB, kr + 40 + kbB);
      v8f s = v8f{};
      s = WMMA_BF16(aq0, b0, s);
      s = WMMA_BF16(aq1, b1, s);
      #pragma unroll
      for (int r = 0; r < 8; ++r) {
        float p = __expf(s[r] - rowM[r]) * rowInvL[r];
        Pl[(r + 8 * hf) * 256 + wave * 32 + sub * 16 + lrow] = (bf16_t)p;
      }
    }
    __syncthreads();
    // --- out GEMM over this 256-wide m chunk; wave owns 64 channels ---
    #pragma unroll
    for (int ct = 0; ct < 4; ++ct) {
      int c = wave * 64 + ct * 16 + lrow;   // A row for this lane
      const bf16_t* vr = vws + ((size_t)b * NC + c) * NHW + mbase;
      // prologue: kk = 0 fragments
      v16bf av = load_frag(vr + kbB, vr + kbB + 8);
      v16bf bp = load_frag(&Pl[lrow * 256 + kbB], &Pl[lrow * 256 + kbB] + 8);
      #pragma unroll
      for (int kk = 0; kk < 8; ++kk) {
        v16bf av_n, bp_n;
        if (kk < 7) {                       // prefetch next K-step
          av_n = load_frag(vr + (kk + 1) * 32 + kbB, vr + (kk + 1) * 32 + kbB + 8);
          const bf16_t* pp = &Pl[lrow * 256 + (kk + 1) * 32 + kbB];
          bp_n = load_frag(pp, pp + 8);
        }
        acc[ct] = WMMA_BF16(av, bp, acc[ct]);
        if (kk < 7) { av = av_n; bp = bp_n; }
      }
    }
    __syncthreads();
  }

  // --- epilogue: y = gamma*out + x ---
  const float* xb = x  + (size_t)b * NC * NHW;
  float*       yb = out + (size_t)b * NC * NHW;
  #pragma unroll
  for (int ct = 0; ct < 4; ++ct) {
    #pragma unroll
    for (int r = 0; r < 8; ++r) {
      int c = wave * 64 + ct * 16 + r + 8 * hf;
      int n = n0 + lrow;
      size_t idx = (size_t)c * NHW + n;
      yb[idx] = gamma * acc[ct][r] + xb[idx];
    }
  }
}

// ---------------------------------------------------------------------------
extern "C" void kernel_launch(void* const* d_in, const int* in_sizes, int n_in,
                              void* d_out, int out_size, void* d_ws, size_t ws_size,
                              hipStream_t stream)
{
  const float* x     = (const float*)d_in[0];
  const float* Wq    = (const float*)d_in[1];
  const float* bq    = (const float*)d_in[2];
  const float* Wk    = (const float*)d_in[3];
  const float* bk    = (const float*)d_in[4];
  const float* Wv    = (const float*)d_in[5];
  const float* bv    = (const float*)d_in[6];
  const float* gamma = (const float*)d_in[7];
  float* y = (float*)d_out;

  // workspace carve (~49 MB)
  char* ws = (char*)d_ws;
  bf16_t* qws = (bf16_t*)ws; ws += (size_t)NB * NHW * ND * sizeof(bf16_t);
  bf16_t* kws = (bf16_t*)ws; ws += (size_t)NB * NHW * ND * sizeof(bf16_t);
  bf16_t* vws = (bf16_t*)ws; ws += (size_t)NB * NC * NHW * sizeof(bf16_t);
  bf16_t* Wb  = (bf16_t*)ws; ws += (size_t)NO * NC * sizeof(bf16_t);
  float*  bb  = (float*)ws;  ws += (size_t)NO * sizeof(float);
  float*  Mg  = (float*)ws;  ws += (size_t)NB * NHW * sizeof(float);
  float*  Lg  = (float*)ws;

  dim3 grid(NHW / 16, NB);
  dim3 blk(256);
  wcvt_kernel<<<dim3(NO * NC / (256 * 8)), blk, 0, stream>>>(Wq, bq, Wk, bk, Wv, bv, Wb, bb);
  proj_kernel<<<grid, blk, 0, stream>>>(x, Wb, bb, qws, kws, vws);
  softmax_stats_kernel<<<grid, blk, 0, stream>>>(qws, kws, Mg, Lg);
  attn_out_kernel<<<grid, blk, 0, stream>>>(x, gamma, qws, kws, vws, Mg, Lg, y);
}
